// Encoder_55740085567712
// MI455X (gfx1250) — matrix operand order repaired and checked by reading, not compiled, not run on hardware
//
#include <hip/hip_runtime.h>
#include <hip/hip_bf16.h>

typedef __attribute__((ext_vector_type(16))) _Float16 v16h;
typedef __attribute__((ext_vector_type(8)))  _Float16 v8h;
typedef __attribute__((ext_vector_type(8)))  float    v8f;

union V16H { v16h v; v8h h[2]; };

#define NB    128
#define SEQ   256
#define UDIM  512
#define T3U   1536
#define HSTR  520   // f16 LDS row stride (520*2B = 1040B, 16B-aligned, bank-skewed)

// ---------------------------------------------------------------------------
// Pre-swizzle a 512x1536 f32 weight matrix into WMMA B-fragment order (f16).
// Fragment (nt,kt): 32 lanes x 16 halves. h=lane/16, N=nt*16+lane%16,
//   K=kt*32+8h+j (j<8), K=kt*32+16+8h+(j-8) (j>=8).
// Byte offset of fragment element: nt*16384 + kt*1024 + lane*32 + j*2.
__global__ void k_pack_b16(const float* __restrict__ W, _Float16* __restrict__ out) {
    int gid = blockIdx.x * 256 + threadIdx.x;           // 96*16*512 = 786432 total
    if (gid >= 96 * 16 * 512) return;
    int j    = gid & 15;
    int lane = (gid >> 4) & 31;
    int tile = gid >> 9;                                // nt*16 + kt
    int kt   = tile & 15;
    int nt   = tile >> 4;
    int k = kt * 32 + (j & 7) + ((j & 8) ? 16 : 0) + ((lane & 16) ? 8 : 0);
    int n = nt * 16 + (lane & 15);
    out[gid] = (_Float16)W[k * T3U + n];
}

// embproj[c][j] = b0[j] + sum_v emb[c][v] * wx_lr[v][j]   (101 x 1536)
__global__ void k_embproj(const float* __restrict__ emb, const float* __restrict__ wx,
                          const float* __restrict__ b0, float* __restrict__ out) {
    int c = blockIdx.y;
    int j = blockIdx.x * 256 + threadIdx.x;
    float acc = b0[j];
    for (int v = 0; v < 256; ++v) acc += emb[c * 256 + v] * wx[v * T3U + j];
    out[c * T3U + j] = acc;
}

__device__ __forceinline__ float sigmoidf_(float x) {
    return 1.0f / (1.0f + __expf(-x));
}

// ---------------------------------------------------------------------------
// LR GRU: 8 persistent blocks, each owns 16 batch rows, 256 sequential steps.
// Per step: rec = h @ wh (16x1536x512 via f16 WMMA), gather xproj from embproj,
// gate math on C-fragments, write x_lr into d_out (stride 1024, cols [0,512)).
// `dz` is 0 at runtime: defeats LICM hoist+spill of the weight fragments.
// B fragments are double-buffered across k-tiles, addressed as one base
// pointer + constant immediate offsets (no address math in the loop).
__global__ __launch_bounds__(512) void k_gru_lr(
    const int* __restrict__ padded, const float* __restrict__ h0g,
    const float* __restrict__ embproj, const _Float16* __restrict__ whp,
    const float* __restrict__ brec_g, float* __restrict__ xout,
    float* __restrict__ hfin, int dz) {
    __shared__ __align__(32) _Float16 h16[16 * HSTR];
    __shared__ int ltok[16];
    __shared__ int lmask[16];

    const int tid = threadIdx.x, lane = tid & 31, w = tid >> 5;
    const int bt = blockIdx.x;
    const int colq = lane & 15;       // col within 16-wide tile / A-row M
    const int hi   = (lane >> 4);     // lane half

    float h[2][8], out_[2][8];
    float brec[3][2];
    for (int g = 0; g < 3; ++g)
        for (int ur = 0; ur < 2; ++ur)
            brec[g][ur] = brec_g[512 * g + 256 * ur + 16 * w + colq];

    for (int ur = 0; ur < 2; ++ur)
        for (int r = 0; r < 8; ++r) {
            int row = r + hi * 8;
            int u = 256 * ur + 16 * w + colq;
            h[ur][r] = h0g[(bt * 16 + row) * UDIM + u];
            out_[ur][r] = 0.0f;
            h16[row * HSTR + u] = (_Float16)h[ur][r];
        }

    for (int t = 0; t < SEQ; ++t) {
        // loop-variant alias of the weight base (dz == 0): defeats LICM+spill
        const char* pB = (const char*)(whp + (size_t)dz * (size_t)t)
                         + (size_t)lane * 32 + (size_t)w * 16384;
        if (tid < 16) {
            ltok[tid]  = padded[(bt * 16 + tid) * SEQ + (SEQ - 1 - t)];
            lmask[tid] = padded[(bt * 16 + tid) * SEQ + t] != 0;
        }
        __syncthreads();

        v8f zero = 0.0f;
        v8f acc[6];                       // f = g*2 + ur
        #pragma unroll
        for (int f = 0; f < 6; ++f) acc[f] = zero;

        v16h bf[2][6];
        #pragma unroll
        for (int f = 0; f < 6; ++f)
            bf[0][f] = *(const v16h*)(pB + ((f >> 1) * 32 + (f & 1) * 16) * 16384);

        #pragma unroll
        for (int kt = 0; kt < 16; ++kt) {
            const int cur = kt & 1, nx = cur ^ 1;
            if (kt < 15) {
                #pragma unroll
                for (int f = 0; f < 6; ++f)
                    bf[nx][f] = *(const v16h*)(pB
                        + ((f >> 1) * 32 + (f & 1) * 16) * 16384 + (kt + 1) * 1024);
            }
            const int base = kt * 32 + hi * 8;
            V16H a;
            a.h[0] = *(const v8h*)&h16[colq * HSTR + base];
            a.h[1] = *(const v8h*)&h16[colq * HSTR + base + 16];
            #pragma unroll
            for (int f = 0; f < 6; ++f)
                acc[f] = __builtin_amdgcn_wmma_f32_16x16x32_f16(
                    false, a.v, false, bf[cur][f], (short)0, acc[f], false, false);
        }
        __syncthreads();   // all A reads done before h16 is overwritten

        for (int ur = 0; ur < 2; ++ur) {
            const int u = 256 * ur + 16 * w + colq;
            for (int r = 0; r < 8; ++r) {
                int row = r + hi * 8;
                const float* ep = embproj + (size_t)ltok[row] * T3U;
                float z  = sigmoidf_(ep[u]        + acc[0 + ur][r] + brec[0][ur]);
                float rr = sigmoidf_(ep[512 + u]  + acc[2 + ur][r] + brec[1][ur]);
                float hh = tanhf(ep[1024 + u] + rr * (acc[4 + ur][r] + brec[2][ur]));
                float hn = z * h[ur][r] + (1.0f - z) * hh;
                if (lmask[row]) { h[ur][r] = hn; out_[ur][r] = hn; }
                xout[((size_t)(bt * 16 + row) * SEQ + t) * 1024 + u] = out_[ur][r];
                h16[row * HSTR + u] = (_Float16)h[ur][r];
            }
        }
    }

    for (int ur = 0; ur < 2; ++ur)
        for (int r = 0; r < 8; ++r) {
            int row = r + hi * 8;
            int u = 256 * ur + 16 * w + colq;
            hfin[(bt * 16 + row) * UDIM + u] = h[ur][r];
        }
}

// ---------------------------------------------------------------------------
// RL GRU with fused xproj_rl = x_lr[:,t,:] @ wx_rl. Two chained WMMA GEMMs per
// step; z/r gates accumulate xp+rec in one fragment, h-gate kept split.
// Per gate-group: one 4-load clause, then 4 WMMAs (pipelined waits).
__global__ __launch_bounds__(512) void k_gru_rl(
    const int* __restrict__ padded, const float* __restrict__ h0g,
    const float* __restrict__ xlr, const _Float16* __restrict__ wxp,
    const _Float16* __restrict__ whp, const float* __restrict__ brl,
    float* __restrict__ xout, float* __restrict__ hfin, int dz) {
    __shared__ __align__(32) _Float16 h16[16 * HSTR];
    __shared__ __align__(32) _Float16 x16[16 * HSTR];
    __shared__ int lmask[16];

    const int tid = threadIdx.x, lane = tid & 31, w = tid >> 5;
    const int bt = blockIdx.x;
    const int colq = lane & 15;
    const int hi   = (lane >> 4);

    float h[2][8], out_[2][8];
    float bsum[2][2], bx2[2], br2[2];
    for (int ur = 0; ur < 2; ++ur) {
        int u = 256 * ur + 16 * w + colq;
        bsum[0][ur] = brl[u]        + brl[T3U + u];
        bsum[1][ur] = brl[512 + u]  + brl[T3U + 512 + u];
        bx2[ur]     = brl[1024 + u];
        br2[ur]     = brl[T3U + 1024 + u];
    }

    for (int ur = 0; ur < 2; ++ur)
        for (int r = 0; r < 8; ++r) {
            int row = r + hi * 8;
            int u = 256 * ur + 16 * w + colq;
            h[ur][r] = h0g[(bt * 16 + row) * UDIM + u];
            out_[ur][r] = 0.0f;
            h16[row * HSTR + u] = (_Float16)h[ur][r];
        }

    for (int t = 0; t < SEQ; ++t) {
        const char* pBx = (const char*)(wxp + (size_t)dz * (size_t)t)
                          + (size_t)lane * 32 + (size_t)w * 16384;
        const char* pBh = (const char*)(whp + (size_t)dz * (size_t)t)
                          + (size_t)lane * 32 + (size_t)w * 16384;
        if (tid < 16)
            lmask[tid] = padded[(bt * 16 + tid) * SEQ + (SEQ - 1 - t)] != 0;
        // stage x_lr[:,t,:] tile as f16
        for (int i = tid; i < 16 * 512; i += 512) {
            int row = i >> 9, c = i & 511;
            x16[row * HSTR + c] =
                (_Float16)xlr[((size_t)(bt * 16 + row) * SEQ + t) * 1024 + c];
        }
        __syncthreads();

        v8f zero = 0.0f;
        v8f s0[2] = {zero, zero}, s1[2] = {zero, zero};
        v8f ax2[2] = {zero, zero}, ah2[2] = {zero, zero};

        #pragma unroll
        for (int kt = 0; kt < 16; ++kt) {
            const int base = kt * 32 + hi * 8;
            V16H ah_, ax_;
            ah_.h[0] = *(const v8h*)&h16[colq * HSTR + base];
            ah_.h[1] = *(const v8h*)&h16[colq * HSTR + base + 16];
            ax_.h[0] = *(const v8h*)&x16[colq * HSTR + base];
            ax_.h[1] = *(const v8h*)&x16[colq * HSTR + base + 16];
            #pragma unroll
            for (int g = 0; g < 3; ++g) {
                const int o0 = (g * 32) * 16384 + kt * 1024;
                const int o1 = (g * 32 + 16) * 16384 + kt * 1024;
                v16h bx0 = *(const v16h*)(pBx + o0);
                v16h bx1 = *(const v16h*)(pBx + o1);
                v16h bh0 = *(const v16h*)(pBh + o0);
                v16h bh1 = *(const v16h*)(pBh + o1);
                if (g == 0) {
                    s0[0] = __builtin_amdgcn_wmma_f32_16x16x32_f16(false, ax_.v, false, bx0, (short)0, s0[0], false, false);
                    s0[0] = __builtin_amdgcn_wmma_f32_16x16x32_f16(false, ah_.v, false, bh0, (short)0, s0[0], false, false);
                    s0[1] = __builtin_amdgcn_wmma_f32_16x16x32_f16(false, ax_.v, false, bx1, (short)0, s0[1], false, false);
                    s0[1] = __builtin_amdgcn_wmma_f32_16x16x32_f16(false, ah_.v, false, bh1, (short)0, s0[1], false, false);
                } else if (g == 1) {
                    s1[0] = __builtin_amdgcn_wmma_f32_16x16x32_f16(false, ax_.v, false, bx0, (short)0, s1[0], false, false);
                    s1[0] = __builtin_amdgcn_wmma_f32_16x16x32_f16(false, ah_.v, false, bh0, (short)0, s1[0], false, false);
                    s1[1] = __builtin_amdgcn_wmma_f32_16x16x32_f16(false, ax_.v, false, bx1, (short)0, s1[1], false, false);
                    s1[1] = __builtin_amdgcn_wmma_f32_16x16x32_f16(false, ah_.v, false, bh1, (short)0, s1[1], false, false);
                } else {
                    ax2[0] = __builtin_amdgcn_wmma_f32_16x16x32_f16(false, ax_.v, false, bx0, (short)0, ax2[0], false, false);
                    ax2[1] = __builtin_amdgcn_wmma_f32_16x16x32_f16(false, ax_.v, false, bx1, (short)0, ax2[1], false, false);
                    ah2[0] = __builtin_amdgcn_wmma_f32_16x16x32_f16(false, ah_.v, false, bh0, (short)0, ah2[0], false, false);
                    ah2[1] = __builtin_amdgcn_wmma_f32_16x16x32_f16(false, ah_.v, false, bh1, (short)0, ah2[1], false, false);
                }
            }
        }
        __syncthreads();

        for (int ur = 0; ur < 2; ++ur) {
            const int u = 256 * ur + 16 * w + colq;
            for (int r = 0; r < 8; ++r) {
                int row = r + hi * 8;
                float z  = sigmoidf_(s0[ur][r] + bsum[0][ur]);
                float rr = sigmoidf_(s1[ur][r] + bsum[1][ur]);
                float hh = tanhf(ax2[ur][r] + bx2[ur] + rr * (ah2[ur][r] + br2[ur]));
                float hn = z * h[ur][r] + (1.0f - z) * hh;
                if (lmask[row]) { h[ur][r] = hn; out_[ur][r] = hn; }
                // x_rl reversed in time, written to cols [512,1024)
                xout[((size_t)(bt * 16 + row) * SEQ + (SEQ - 1 - t)) * 1024 + 512 + u] = out_[ur][r];
                h16[row * HSTR + u] = (_Float16)h[ur][r];
            }
        }
    }

    for (int ur = 0; ur < 2; ++ur)
        for (int r = 0; r < 8; ++r) {
            int row = r + hi * 8;
            int u = 256 * ur + 16 * w + colq;
            hfin[(bt * 16 + row) * UDIM + u] = h[ur][r];
        }
}

// ---------------------------------------------------------------------------
// Convs k=2..5 over x_lr with relu, mask penalty, and max-over-time.
// One block per batch row; x window staged f16 in LDS per 32-step chunk.
__global__ __launch_bounds__(256) void k_conv(
    const float* __restrict__ xlr, const int* __restrict__ padded,
    const float* __restrict__ w2, const float* __restrict__ b2,
    const float* __restrict__ w3, const float* __restrict__ b3,
    const float* __restrict__ w4, const float* __restrict__ b4,
    const float* __restrict__ w5, const float* __restrict__ b5,
    float* __restrict__ cnn) {
    __shared__ _Float16 xw[36 * 512];
    __shared__ float red[4][64];
    const int b = blockIdx.x;
    const int tid = threadIdx.x;
    const int fi = tid & 63;   // global filter index 0..59 (60..63 idle)
    const int tg = tid >> 6;   // time group 0..3 within chunk

    int k = 2, fbase = 0, fcnt = 4;
    const float *wp = w2, *bp = b2;
    if (fi >= 28)      { k = 5; fbase = 28; fcnt = 32; wp = w5; bp = b5; }
    else if (fi >= 12) { k = 4; fbase = 12; fcnt = 16; wp = w4; bp = b4; }
    else if (fi >= 4)  { k = 3; fbase = 4;  fcnt = 8;  wp = w3; bp = b3; }
    const int fl = fi - fbase;

    float fmax = -INFINITY;
    for (int tc = 0; tc < 8; ++tc) {
        const int t0 = tc * 32;
        __syncthreads();
        for (int i = tid; i < 36 * 512; i += 256) {
            int row = i >> 9, c = i & 511;
            int gt = t0 + row;
            float v = (gt < SEQ) ? xlr[((size_t)b * SEQ + gt) * 1024 + c] : 0.0f;
            xw[i] = (_Float16)v;
        }
        __syncthreads();
        if (fi < 60) {
            float acc[8];
            for (int tt = 0; tt < 8; ++tt) acc[tt] = bp[fl];
            for (int dt = 0; dt < k; ++dt)
                for (int c = 0; c < 512; ++c) {
                    float wv = wp[(dt * 512 + c) * fcnt + fl];
                    const _Float16* xr = &xw[(tg * 8 + dt) * 512 + c];
                    for (int tt = 0; tt < 8; ++tt)
                        acc[tt] += (float)xr[tt * 512] * wv;
                }
            for (int tt = 0; tt < 8; ++tt) {
                int t = t0 + tg * 8 + tt;
                if (t <= SEQ - k) {
                    float v = acc[tt] > 0.0f ? acc[tt] : 0.0f;
                    if (padded[b * SEQ + t + k - 1] != 0) v -= 1.0e9f;
                    if (v > fmax) fmax = v;
                }
            }
        }
    }
    red[tg][fi] = fmax;
    __syncthreads();
    if (tid < 60) {
        float m = red[0][tid];
        m = fmaxf(m, red[1][tid]);
        m = fmaxf(m, red[2][tid]);
        m = fmaxf(m, red[3][tid]);
        cnn[b * 64 + tid] = m;
    }
}

// g = [h_lr | h_rl | cnn] @ proj_w + proj_b
__global__ void k_proj(const float* __restrict__ hlr, const float* __restrict__ hrl,
                       const float* __restrict__ cnn, const float* __restrict__ pw,
                       const float* __restrict__ pb, float* __restrict__ out) {
    int b = blockIdx.y;
    int u = blockIdx.x * 256 + threadIdx.x;
    float acc = pb[u];
    for (int i = 0; i < 512; ++i) acc += hlr[b * 512 + i] * pw[i * 512 + u];
    for (int i = 0; i < 512; ++i) acc += hrl[b * 512 + i] * pw[(512 + i) * 512 + u];
    for (int j = 0; j < 60; ++j)  acc += cnn[b * 64 + j] * pw[(1024 + j) * 512 + u];
    out[b * 512 + u] = acc;
}

// ---------------------------------------------------------------------------
extern "C" void kernel_launch(void* const* d_in, const int* in_sizes, int n_in,
                              void* d_out, int out_size, void* d_ws, size_t ws_size,
                              hipStream_t stream) {
    const int*   padded  = (const int*)d_in[0];
    const float* hidden  = (const float*)d_in[1];
    const float* emb     = (const float*)d_in[2];
    const float* wx_lr   = (const float*)d_in[3];
    const float* wh_lr   = (const float*)d_in[4];
    const float* b_lr    = (const float*)d_in[5];
    const float* wx_rl   = (const float*)d_in[6];
    const float* wh_rl   = (const float*)d_in[7];
    const float* b_rl    = (const float*)d_in[8];
    const float* w2 = (const float*)d_in[9];  const float* b2 = (const float*)d_in[10];
    const float* w3 = (const float*)d_in[11]; const float* b3 = (const float*)d_in[12];
    const float* w4 = (const float*)d_in[13]; const float* b4 = (const float*)d_in[14];
    const float* w5 = (const float*)d_in[15]; const float* b5 = (const float*)d_in[16];
    const float* pw = (const float*)d_in[17]; const float* pb = (const float*)d_in[18];

    char* ws = (char*)d_ws;
    float*    embproj = (float*)(ws + 0);                 // 101*1536 f32 = 620,544 B
    _Float16* whlr16  = (_Float16*)(ws + 620544);         // 1,572,864 B each
    _Float16* whrl16  = (_Float16*)(ws + 2193408);
    _Float16* wxrl16  = (_Float16*)(ws + 3766272);
    float*    hlr     = (float*)(ws + 5339136);           // 262,144 B
    float*    hrl     = (float*)(ws + 5601280);           // 262,144 B
    float*    cnn     = (float*)(ws + 5863424);           // 32,768 B

    float* x    = (float*)d_out;                          // (B,S,1024)
    float* proj = (float*)d_out + (size_t)NB * SEQ * 1024;

    k_pack_b16<<<3072, 256, 0, stream>>>(wh_lr, whlr16);
    k_pack_b16<<<3072, 256, 0, stream>>>(wh_rl, whrl16);
    k_pack_b16<<<3072, 256, 0, stream>>>(wx_rl, wxrl16);
    k_embproj<<<dim3(6, 101), 256, 0, stream>>>(emb, wx_lr, b_lr, embproj);

    k_gru_lr<<<8, 512, 0, stream>>>(padded, hidden, embproj, whlr16,
                                    b_lr + T3U, x, hlr, /*dz=*/0);
    k_gru_rl<<<8, 512, 0, stream>>>(padded, hidden, x, wxrl16, whrl16,
                                    b_rl, x, hrl, /*dz=*/0);
    k_conv<<<NB, 256, 0, stream>>>(x, padded, w2, b2, w3, b3, w4, b4, w5, b5, cnn);
    k_proj<<<dim3(2, NB), 256, 0, stream>>>(hlr, hrl, cnn, pw, pb, proj);
}
